// MixtureOfExperts_80169859548041
// MI455X (gfx1250) — matrix operand order repaired and checked by reading, not compiled, not run on hardware
//
#include <hip/hip_runtime.h>
#include <math.h>

// Problem constants (B=2, S=1024 -> T tokens)
#define T_TOK 2048
#define D_DIM 1024
#define F_DIM 4096
#define E_EXP 8

typedef __attribute__((ext_vector_type(16))) __bf16 v16bf;
typedef __attribute__((ext_vector_type(8)))  __bf16 v8bf;
typedef __attribute__((ext_vector_type(8)))  float  v8f;

#define LDSS 40                      // padded LDS row stride (elements)
#define TILE_ELEMS (128 * LDSS)      // one 128x32 staged tile (padded)
#define TILE_BYTES (TILE_ELEMS * 2)

// ---------------- workspace layout (bytes) ----------------
#define OFF_W1T  ((size_t)0)                         // [E][F][D] bf16  = 64 MiB
#define OFF_W2T  ((size_t)67108864)                  // [E][D][F] bf16  = 64 MiB
#define OFF_XB   ((size_t)134217728)                 // [T][D]    bf16  =  4 MiB
#define OFF_H    ((size_t)138412032)                 // [4096][F] bf16  = 32 MiB
#define OFF_META ((size_t)171966464)

// ---------------- CDNA5 async global->LDS copy (16B per lane) ----------------
__device__ __forceinline__ void async_copy16(unsigned lds_off, unsigned long long gaddr) {
  asm volatile("global_load_async_to_lds_b128 %0, %1, off"
               :: "v"(lds_off), "v"(gaddr) : "memory");
}
__device__ __forceinline__ void wait_async0() {
  asm volatile("s_wait_asynccnt 0" ::: "memory");
}

// ---------------- WMMA fragment loads from LDS ----------------
// A (16x32 bf16, MxK): lane<16 -> M=lane, K={0..7,16..23}; lane>=16 -> K={8..15,24..31}
__device__ __forceinline__ v16bf frag_a(const __bf16* lds, int row, int lhi) {
  const __bf16* p = lds + row * LDSS + lhi * 8;
  v8bf lo = *(const v8bf*)p;
  v8bf hi = *(const v8bf*)(p + 16);
  return __builtin_shufflevector(lo, hi, 0,1,2,3,4,5,6,7,8,9,10,11,12,13,14,15);
}
// B (32x16 bf16, KxN): lane<16 -> N=lane, K={0..7,16..23}; lane>=16 -> K={8..15,24..31}
__device__ __forceinline__ v16bf frag_b(const __bf16* lds, int row, int lhi) {
  const __bf16* p = lds + row * LDSS + lhi * 8;
  v8bf lo = *(const v8bf*)p;
  v8bf hi = *(const v8bf*)(p + 16);
  return __builtin_shufflevector(lo, hi, 0,1,2,3,4,5,6,7,8,9,10,11,12,13,14,15);
}

// ---------------- transpose + fp32 -> bf16 convert ----------------
__global__ __launch_bounds__(256) void moe_transpose(
    const float* __restrict__ in, __bf16* __restrict__ out, int R, int C) {
  const float* src = in  + (size_t)blockIdx.z * R * C;
  __bf16*      dst = out + (size_t)blockIdx.z * R * C;
  __shared__ float tile[32][33];
  int tx = threadIdx.x, ty = threadIdx.y;
  int r0 = blockIdx.y * 32, c0 = blockIdx.x * 32;
#pragma unroll
  for (int i = 0; i < 4; ++i)
    tile[ty + 8 * i][tx] = src[(size_t)(r0 + ty + 8 * i) * C + c0 + tx];
  __syncthreads();
#pragma unroll
  for (int i = 0; i < 4; ++i)
    dst[(size_t)(c0 + ty + 8 * i) * R + r0 + tx] = (__bf16)tile[tx][ty + 8 * i];
}

// ---------------- router ----------------
__global__ __launch_bounds__(256) void moe_router(
    const float* __restrict__ x, const float* __restrict__ Wr,
    const float* __restrict__ br, __bf16* __restrict__ XB,
    float* __restrict__ probsOut, int* __restrict__ counts,
    int* __restrict__ tokE, float* __restrict__ tokP) {
  const int t = blockIdx.x;
  const int tid = threadIdx.x;
  const int lane = tid & 31, wave = tid >> 5;  // 8 waves = 8 experts
  __shared__ float logits[E_EXP];
  const float* xr = x + (size_t)t * D_DIM;

  for (int d = tid; d < D_DIM; d += 256) XB[(size_t)t * D_DIM + d] = (__bf16)xr[d];

  float acc = 0.f;
  for (int d = lane; d < D_DIM; d += 32) acc += xr[d] * Wr[d * E_EXP + wave];
#pragma unroll
  for (int off = 16; off > 0; off >>= 1) acc += __shfl_down(acc, off, 32);
  if (lane == 0) logits[wave] = acc + br[wave];
  __syncthreads();

  if (tid == 0) {
    float l[E_EXP];
#pragma unroll
    for (int e = 0; e < E_EXP; ++e) l[e] = logits[e];
    int i0 = 0;
    for (int e = 1; e < E_EXP; ++e) if (l[e] > l[i0]) i0 = e;  // ties -> lower idx
    int i1 = -1;
    for (int e = 0; e < E_EXP; ++e) {
      if (e == i0) continue;
      if (i1 < 0 || l[e] > l[i1]) i1 = e;
    }
    float e1 = expf(l[i1] - l[i0]);
    float p0 = 1.0f / (1.0f + e1), p1 = e1 / (1.0f + e1);
    probsOut[t * 2] = p0;  probsOut[t * 2 + 1] = p1;
    tokE[t * 2] = i0;      tokE[t * 2 + 1] = i1;
    tokP[t * 2] = p0;      tokP[t * 2 + 1] = p1;
    atomicAdd(&counts[i0], 1);
    atomicAdd(&counts[i1], 1);
  }
}

__global__ void moe_scan(const int* __restrict__ counts, int* __restrict__ offsets) {
  if (threadIdx.x == 0 && blockIdx.x == 0) {
    int s = 0;
    for (int e = 0; e < E_EXP; ++e) { offsets[e] = s; s += counts[e]; }
  }
}

__global__ __launch_bounds__(256) void moe_scatter(
    const int* __restrict__ tokE, const float* __restrict__ tokP,
    const int* __restrict__ offsets, int* __restrict__ cnt2,
    int* __restrict__ aTok, float* __restrict__ aW) {
  int t = blockIdx.x * blockDim.x + threadIdx.x;
  if (t >= T_TOK) return;
#pragma unroll
  for (int k = 0; k < 2; ++k) {
    int e = tokE[t * 2 + k];
    int pos = atomicAdd(&cnt2[e], 1);
    int row = offsets[e] + pos;
    aTok[row] = t;
    aW[row] = tokP[t * 2 + k];
  }
}

// ---------------- grouped GEMM1: h^T = W1^T x^T, GELU, store h bf16 ----------------
// block tile M(F)=128 x N(tok)=128; 8 waves as 4x2; wave tile 32x64 (2x4 WMMA frags)
// double-buffered LDS, async global->LDS staging, one barrier per K-step
__global__ __launch_bounds__(256) void moe_gemm1(
    const __bf16* __restrict__ W1T, const __bf16* __restrict__ XB,
    const float* __restrict__ b1, const int* __restrict__ counts,
    const int* __restrict__ offsets, const int* __restrict__ aTok,
    __bf16* __restrict__ H) {
  const int e = blockIdx.z;
  const int cnt = counts[e];
  const int ntile = blockIdx.y * 128;
  if (ntile >= cnt) return;
  const int base = offsets[e];
  const int ftile = blockIdx.x * 128;

  __shared__ __bf16 ldsA[2][TILE_ELEMS];
  __shared__ __bf16 ldsB[2][TILE_ELEMS];
  __shared__ int tokIdx[128];

  const int tid = threadIdx.x;
  if (tid < 128) {
    int p = ntile + tid;
    tokIdx[tid] = aTok[base + (p < cnt ? p : cnt - 1)];
  }
  __syncthreads();

  const unsigned ldsA0 = (unsigned)(size_t)&ldsA[0][0];  // low 32 bits = LDS offset
  const unsigned ldsB0 = (unsigned)(size_t)&ldsB[0][0];
  const int lane = tid & 31, wave = tid >> 5;
  const int wm = (wave >> 1) * 32;   // 4 waves along M
  const int wn = (wave & 1) * 64;    // 2 waves along N
  const int lhi = lane >> 4, lm = lane & 15;

  // staging coords: each thread moves 2 A-chunks + 2 B-chunks of 8 bf16
  const int m0 = tid >> 2, c0 = (tid & 3) * 8;   // rows 0..63
  const int m1 = m0 + 64;                        // rows 64..127
  const __bf16* Wb = W1T + (size_t)e * F_DIM * D_DIM + (size_t)ftile * D_DIM;
  const __bf16* Ar0 = Wb + (size_t)m0 * D_DIM + c0;
  const __bf16* Ar1 = Wb + (size_t)m1 * D_DIM + c0;
  const __bf16* Br0 = XB + (size_t)tokIdx[m0] * D_DIM + c0;
  const __bf16* Br1 = XB + (size_t)tokIdx[m1] * D_DIM + c0;
  const unsigned la0 = ldsA0 + (unsigned)(m0 * LDSS + c0) * 2;
  const unsigned la1 = ldsA0 + (unsigned)(m1 * LDSS + c0) * 2;
  const unsigned lb0 = ldsB0 + (unsigned)(m0 * LDSS + c0) * 2;
  const unsigned lb1 = ldsB0 + (unsigned)(m1 * LDSS + c0) * 2;

  v8f acc[2][4] = {};
  const int NT = D_DIM / 32;

  // prologue: tile 0 -> buffer 0
  async_copy16(la0, (unsigned long long)(size_t)Ar0);
  async_copy16(la1, (unsigned long long)(size_t)Ar1);
  async_copy16(lb0, (unsigned long long)(size_t)Br0);
  async_copy16(lb1, (unsigned long long)(size_t)Br1);

  for (int i = 0; i < NT; ++i) {
    wait_async0();        // this wave's portion of tile i is in LDS
    __syncthreads();      // everyone's tile i visible; buffers from i-1 free
    if (i + 1 < NT) {     // async-fill tile i+1 while computing tile i
      const int kk = (i + 1) * 32;
      const unsigned bo = (unsigned)(((i + 1) & 1) * TILE_BYTES);
      async_copy16(la0 + bo, (unsigned long long)(size_t)(Ar0 + kk));
      async_copy16(la1 + bo, (unsigned long long)(size_t)(Ar1 + kk));
      async_copy16(lb0 + bo, (unsigned long long)(size_t)(Br0 + kk));
      async_copy16(lb1 + bo, (unsigned long long)(size_t)(Br1 + kk));
    }
    const __bf16* As = ldsA[i & 1];
    const __bf16* Bs = ldsB[i & 1];
    v16bf a0 = frag_a(As, wm + lm, lhi);
    v16bf a1 = frag_a(As, wm + 16 + lm, lhi);
    v16bf bb0 = frag_b(Bs, wn + lm, lhi);
    v16bf bb1 = frag_b(Bs, wn + 16 + lm, lhi);
    v16bf bb2 = frag_b(Bs, wn + 32 + lm, lhi);
    v16bf bb3 = frag_b(Bs, wn + 48 + lm, lhi);
    acc[0][0] = __builtin_amdgcn_wmma_f32_16x16x32_bf16(false, a0, false, bb0, (short)0, acc[0][0], false, false);
    acc[0][1] = __builtin_amdgcn_wmma_f32_16x16x32_bf16(false, a0, false, bb1, (short)0, acc[0][1], false, false);
    acc[0][2] = __builtin_amdgcn_wmma_f32_16x16x32_bf16(false, a0, false, bb2, (short)0, acc[0][2], false, false);
    acc[0][3] = __builtin_amdgcn_wmma_f32_16x16x32_bf16(false, a0, false, bb3, (short)0, acc[0][3], false, false);
    acc[1][0] = __builtin_amdgcn_wmma_f32_16x16x32_bf16(false, a1, false, bb0, (short)0, acc[1][0], false, false);
    acc[1][1] = __builtin_amdgcn_wmma_f32_16x16x32_bf16(false, a1, false, bb1, (short)0, acc[1][1], false, false);
    acc[1][2] = __builtin_amdgcn_wmma_f32_16x16x32_bf16(false, a1, false, bb2, (short)0, acc[1][2], false, false);
    acc[1][3] = __builtin_amdgcn_wmma_f32_16x16x32_bf16(false, a1, false, bb3, (short)0, acc[1][3], false, false);
  }

#pragma unroll
  for (int mi = 0; mi < 2; ++mi)
#pragma unroll
    for (int ni = 0; ni < 4; ++ni) {
      int n = wn + ni * 16 + lm;
      int pos = ntile + n;
      if (pos >= cnt) continue;  // don't pollute neighbor experts' h rows
      int f0 = ftile + wm + mi * 16 + lhi * 8;
      const float* bp = b1 + (size_t)e * F_DIM + f0;
      v8bf o;
#pragma unroll
      for (int r = 0; r < 8; ++r) {
        float v = acc[mi][ni][r] + bp[r];
        v = 0.5f * v * (1.0f + erff(v * 0.70710678118654752f));  // exact GELU
        o[r] = (__bf16)v;
      }
      *(v8bf*)&H[(size_t)(base + pos) * F_DIM + f0] = o;
    }
}

// ---------------- grouped GEMM2: out^T += w * (W2^T h^T + b2) ----------------
__global__ __launch_bounds__(256) void moe_gemm2(
    const __bf16* __restrict__ W2T, const __bf16* __restrict__ H,
    const float* __restrict__ b2, const int* __restrict__ counts,
    const int* __restrict__ offsets, const int* __restrict__ aTok,
    const float* __restrict__ aW, float* __restrict__ OUT) {
  const int e = blockIdx.z;
  const int cnt = counts[e];
  const int ntile = blockIdx.y * 128;
  if (ntile >= cnt) return;
  const int base = offsets[e];
  const int dtile = blockIdx.x * 128;

  __shared__ __bf16 ldsA[2][TILE_ELEMS];
  __shared__ __bf16 ldsB[2][TILE_ELEMS];
  __shared__ int rowIdx[128];

  const int tid = threadIdx.x;
  if (tid < 128) {
    int p = ntile + tid;
    rowIdx[tid] = base + (p < cnt ? p : cnt - 1);
  }
  __syncthreads();

  const unsigned ldsA0 = (unsigned)(size_t)&ldsA[0][0];
  const unsigned ldsB0 = (unsigned)(size_t)&ldsB[0][0];
  const int lane = tid & 31, wave = tid >> 5;
  const int wm = (wave >> 1) * 32;
  const int wn = (wave & 1) * 64;
  const int lhi = lane >> 4, lm = lane & 15;

  const int m0 = tid >> 2, c0 = (tid & 3) * 8;
  const int m1 = m0 + 64;
  const __bf16* Wb = W2T + (size_t)e * D_DIM * F_DIM + (size_t)dtile * F_DIM;
  const __bf16* Ar0 = Wb + (size_t)m0 * F_DIM + c0;
  const __bf16* Ar1 = Wb + (size_t)m1 * F_DIM + c0;
  const __bf16* Br0 = H + (size_t)rowIdx[m0] * F_DIM + c0;
  const __bf16* Br1 = H + (size_t)rowIdx[m1] * F_DIM + c0;
  const unsigned la0 = ldsA0 + (unsigned)(m0 * LDSS + c0) * 2;
  const unsigned la1 = ldsA0 + (unsigned)(m1 * LDSS + c0) * 2;
  const unsigned lb0 = ldsB0 + (unsigned)(m0 * LDSS + c0) * 2;
  const unsigned lb1 = ldsB0 + (unsigned)(m1 * LDSS + c0) * 2;

  v8f acc[2][4] = {};
  const int NT = F_DIM / 32;

  async_copy16(la0, (unsigned long long)(size_t)Ar0);
  async_copy16(la1, (unsigned long long)(size_t)Ar1);
  async_copy16(lb0, (unsigned long long)(size_t)Br0);
  async_copy16(lb1, (unsigned long long)(size_t)Br1);

  for (int i = 0; i < NT; ++i) {
    wait_async0();
    __syncthreads();
    if (i + 1 < NT) {
      const int kk = (i + 1) * 32;
      const unsigned bo = (unsigned)(((i + 1) & 1) * TILE_BYTES);
      async_copy16(la0 + bo, (unsigned long long)(size_t)(Ar0 + kk));
      async_copy16(la1 + bo, (unsigned long long)(size_t)(Ar1 + kk));
      async_copy16(lb0 + bo, (unsigned long long)(size_t)(Br0 + kk));
      async_copy16(lb1 + bo, (unsigned long long)(size_t)(Br1 + kk));
    }
    const __bf16* As = ldsA[i & 1];
    const __bf16* Bs = ldsB[i & 1];
    v16bf a0 = frag_a(As, wm + lm, lhi);
    v16bf a1 = frag_a(As, wm + 16 + lm, lhi);
    v16bf bb0 = frag_b(Bs, wn + lm, lhi);
    v16bf bb1 = frag_b(Bs, wn + 16 + lm, lhi);
    v16bf bb2 = frag_b(Bs, wn + 32 + lm, lhi);
    v16bf bb3 = frag_b(Bs, wn + 48 + lm, lhi);
    acc[0][0] = __builtin_amdgcn_wmma_f32_16x16x32_bf16(false, a0, false, bb0, (short)0, acc[0][0], false, false);
    acc[0][1] = __builtin_amdgcn_wmma_f32_16x16x32_bf16(false, a0, false, bb1, (short)0, acc[0][1], false, false);
    acc[0][2] = __builtin_amdgcn_wmma_f32_16x16x32_bf16(false, a0, false, bb2, (short)0, acc[0][2], false, false);
    acc[0][3] = __builtin_amdgcn_wmma_f32_16x16x32_bf16(false, a0, false, bb3, (short)0, acc[0][3], false, false);
    acc[1][0] = __builtin_amdgcn_wmma_f32_16x16x32_bf16(false, a1, false, bb0, (short)0, acc[1][0], false, false);
    acc[1][1] = __builtin_amdgcn_wmma_f32_16x16x32_bf16(false, a1, false, bb1, (short)0, acc[1][1], false, false);
    acc[1][2] = __builtin_amdgcn_wmma_f32_16x16x32_bf16(false, a1, false, bb2, (short)0, acc[1][2], false, false);
    acc[1][3] = __builtin_amdgcn_wmma_f32_16x16x32_bf16(false, a1, false, bb3, (short)0, acc[1][3], false, false);
  }

#pragma unroll
  for (int mi = 0; mi < 2; ++mi)
#pragma unroll
    for (int ni = 0; ni < 4; ++ni) {
      int n = wn + ni * 16 + lm;
      int pos = ntile + n;
      if (pos >= cnt) continue;
      int row = base + pos;
      int tok = aTok[row];
      float w = aW[row];
      int d0 = dtile + wm + mi * 16 + lhi * 8;
      const float* bp = b2 + (size_t)e * D_DIM + d0;
      float* op = OUT + (size_t)tok * D_DIM + d0;
#pragma unroll
      for (int r = 0; r < 8; ++r)
        unsafeAtomicAdd(&op[r], w * (acc[mi][ni][r] + bp[r]));  // exactly 2 adders/elem
    }
}

// ---------------- host-side launch ----------------
extern "C" void kernel_launch(void* const* d_in, const int* in_sizes, int n_in,
                              void* d_out, int out_size, void* d_ws, size_t ws_size,
                              hipStream_t stream) {
  const float* x  = (const float*)d_in[0];
  const float* Wr = (const float*)d_in[1];
  const float* br = (const float*)d_in[2];
  const float* W1 = (const float*)d_in[3];
  const float* b1 = (const float*)d_in[4];
  const float* W2 = (const float*)d_in[5];
  const float* b2 = (const float*)d_in[6];
  float* out = (float*)d_out;

  char* ws = (char*)d_ws;
  __bf16* W1T  = (__bf16*)(ws + OFF_W1T);
  __bf16* W2T  = (__bf16*)(ws + OFF_W2T);
  __bf16* XB   = (__bf16*)(ws + OFF_XB);
  __bf16* H    = (__bf16*)(ws + OFF_H);
  int*    counts  = (int*)  (ws + OFF_META + 0);
  int*    offsets = (int*)  (ws + OFF_META + 32);
  int*    cnt2    = (int*)  (ws + OFF_META + 64);
  int*    tokE    = (int*)  (ws + OFF_META + 128);
  float*  tokP    = (float*)(ws + OFF_META + 16512);
  int*    aTok    = (int*)  (ws + OFF_META + 32896);
  float*  aW      = (float*)(ws + OFF_META + 49280);

  hipMemsetAsync(ws + OFF_META, 0, 128, stream);  // counters
  hipMemsetAsync(d_out, 0, (size_t)T_TOK * D_DIM * sizeof(float), stream);

  dim3 tb(32, 8);
  moe_transpose<<<dim3(F_DIM / 32, D_DIM / 32, E_EXP), tb, 0, stream>>>(W1, W1T, D_DIM, F_DIM);
  moe_transpose<<<dim3(D_DIM / 32, F_DIM / 32, E_EXP), tb, 0, stream>>>(W2, W2T, F_DIM, D_DIM);

  moe_router<<<T_TOK, 256, 0, stream>>>(x, Wr, br, XB,
                                        out + (size_t)T_TOK * D_DIM,  // probs tail
                                        counts, tokE, tokP);
  moe_scan<<<1, 32, 0, stream>>>(counts, offsets);
  moe_scatter<<<T_TOK / 256, 256, 0, stream>>>(tokE, tokP, offsets, cnt2, aTok, aW);

  moe_gemm1<<<dim3(F_DIM / 128, T_TOK / 128, E_EXP), 256, 0, stream>>>(
      W1T, XB, b1, counts, offsets, aTok, H);
  moe_gemm2<<<dim3(D_DIM / 128, T_TOK / 128, E_EXP), 256, 0, stream>>>(
      W2T, H, b2, counts, offsets, aTok, aW, out);
}
